// deform_block_middle_22797686407587
// MI455X (gfx1250) — matrix operand order repaired and checked by reading, not compiled, not run on hardware
//
// MI455X (gfx1250, wave32) implementation of the deform/edge-conv block.
// Heavy GEMMs -> v_wmma_f32_16x16x32_bf16 (fp32 accum). Gathers fused into
// WMMA B-operand loads. Workspace region0 (84MB) is reused: dist -> ww.
#include <hip/hip_runtime.h>
#include <hip/hip_bf16.h>

#define BB 4
#define CC 128
#define NN 2048
#define KK 20

typedef __bf16 bf16_t;
typedef __bf16 v16bf __attribute__((ext_vector_type(16)));
typedef float  v8f   __attribute__((ext_vector_type(8)));

__device__ __forceinline__ float lrelu(float v) { return v >= 0.f ? v : 0.01f * v; }
// 16-bit A-matrix 16x32 layout: elem j of lane (half = lane>>4) -> K offset
__device__ __forceinline__ int a_koff(int j, int half) { return ((j >> 3) << 4) + (half << 3) + (j & 7); }
// 16-bit B-matrix 32x16 layout: elem j -> K offset
__device__ __forceinline__ int b_koff(int j, int half) { return ((j >> 3) << 4) + (half << 3) + (j & 7); }

// ---------------- prep: x2 = sum_c x^2, xbf = bf16(x) ----------------
__global__ void prep_kernel(const float* __restrict__ x, bf16_t* __restrict__ xbf, float* __restrict__ x2) {
  int t = blockIdx.x * blockDim.x + threadIdx.x;      // b*NN + n
  if (t >= BB * NN) return;
  int b = t / NN, n = t % NN;
  const float* xb = x + (size_t)b * CC * NN + n;
  bf16_t* ob = xbf + (size_t)b * CC * NN + n;
  float s = 0.f;
  for (int c = 0; c < CC; c++) { float v = xb[(size_t)c * NN]; s += v * v; ob[(size_t)c * NN] = (bf16_t)v; }
  x2[t] = s;
}

// ---------------- convert weights to bf16 ----------------
__global__ void convw_kernel(const float* __restrict__ w_fea, const float* __restrict__ w_inte,
                             const float* __restrict__ w_c2, const float* __restrict__ w_add,
                             bf16_t* wfeabf, bf16_t* wintebf, bf16_t* w2bf, bf16_t* waddbf) {
  int i = blockIdx.x * blockDim.x + threadIdx.x;
  if (i < 4096)  { wfeabf[i]  = (bf16_t)w_fea[i];  return; } i -= 4096;
  if (i < 32768) { wintebf[i] = (bf16_t)w_inte[i]; return; } i -= 32768;
  if (i < 327680){ w2bf[i]    = (bf16_t)w_c2[i];   return; } i -= 327680;
  if (i < 16384) { int o = i >> 7, c = i & 127; waddbf[i] = (bf16_t)w_add[o * 2 * CC + CC + c]; }
}

// ---------------- dist = -2*Gram + x2_i + x2_j (WMMA bf16) ----------------
__global__ void __launch_bounds__(256) dist_kernel(const bf16_t* __restrict__ xbf,
                                                   const float* __restrict__ x2, float* __restrict__ dist) {
  int wave = threadIdx.x >> 5, lane = threadIdx.x & 31;
  int half = lane >> 4, lm = lane & 15;
  int jt = blockIdx.x * 8 + wave;
  int i0 = blockIdx.y * 16, j0 = jt * 16, b = blockIdx.z;
  const bf16_t* xb = xbf + (size_t)b * CC * NN;
  v8f acc = {};
  for (int k0 = 0; k0 < CC; k0 += 32) {
    v16bf a, bb;
#pragma unroll
    for (int j = 0; j < 16; j++) {
      a[j]  = xb[(size_t)(k0 + a_koff(j, half)) * NN + i0 + lm];
      bb[j] = xb[(size_t)(k0 + b_koff(j, half)) * NN + j0 + lm];
    }
    acc = __builtin_amdgcn_wmma_f32_16x16x32_bf16(false, a, false, bb, (short)0, acc, false, false);
  }
  const float* x2b = x2 + b * NN;
#pragma unroll
  for (int r = 0; r < 8; r++) {
    int i = i0 + r + half * 8, j = j0 + lm;
    dist[((size_t)b * NN + i) * NN + j] = -2.0f * acc[r] + x2b[i] + x2b[j];
  }
}

// ---------------- top-20 nearest (excluding self) per row ----------------
__global__ void topk_kernel(const float* __restrict__ dist, int* __restrict__ idx) {
  int t = blockIdx.x * blockDim.x + threadIdx.x;     // b*NN + i
  if (t >= BB * NN) return;
  const float* row = dist + (size_t)t * NN;
  int i = t % NN;
  float bd[KK]; int bi_[KK];
#pragma unroll
  for (int k = 0; k < KK; k++) { bd[k] = 3.4e38f; bi_[k] = 0; }
  for (int j = 0; j < NN; j++) {
    if (j == i) continue;
    float d = row[j];
    if (d < bd[KK - 1]) {
      int p = KK - 1;
      while (p > 0 && bd[p - 1] > d) { bd[p] = bd[p - 1]; bi_[p] = bi_[p - 1]; p--; }
      bd[p] = d; bi_[p] = j;
    }
  }
  for (int k = 0; k < KK; k++) idx[(size_t)t * KK + k] = bi_[k];
}

// ------- fused w_fea(16x256) + w_inte(128x256) over gathered e_fea columns (WMMA) -------
// columns = (n,k) pairs; B fragments gathered once, reused for 9 row tiles.
__global__ void __launch_bounds__(256) gather_wmma_kernel(const float* __restrict__ x, const int* __restrict__ idx,
                                                          const bf16_t* __restrict__ wfeabf,
                                                          const bf16_t* __restrict__ wintebf,
                                                          float* __restrict__ wfpre, bf16_t* __restrict__ intebf) {
  int wave = threadIdx.x >> 5, lane = threadIdx.x & 31;
  int half = lane >> 4, lm = lane & 15;
  int ct = blockIdx.x * 8 + wave;                 // col tile in [0, NN*KK/16)
  int b = blockIdx.y;
  int col = ct * 16 + lm;
  int n = col / KK, k = col % KK;
  int nb = idx[((size_t)b * NN + n) * KK + k];
  const float* xb = x + (size_t)b * CC * NN;
  v16bf bfrag[8];
#pragma unroll
  for (int s = 0; s < 8; s++) {
    int k0 = s * 32;
#pragma unroll
    for (int j = 0; j < 16; j++) {
      int c = k0 + b_koff(j, half);
      float v;
      if (c < CC) v = xb[(size_t)c * NN + n];
      else { int cc = c - CC; v = xb[(size_t)cc * NN + nb] - xb[(size_t)cc * NN + n]; }
      bfrag[s][j] = (bf16_t)v;
    }
  }
  { // row tile 0: w_fea -> wfpre (B,16,N,K) f32 (pre-activation)
    v8f acc = {};
#pragma unroll
    for (int s = 0; s < 8; s++) {
      v16bf a;
#pragma unroll
      for (int j = 0; j < 16; j++) a[j] = wfeabf[lm * 256 + s * 32 + a_koff(j, half)];
      acc = __builtin_amdgcn_wmma_f32_16x16x32_bf16(false, a, false, bfrag[s], (short)0, acc, false, false);
    }
#pragma unroll
    for (int r = 0; r < 8; r++) {
      int m = r + half * 8;
      wfpre[(((size_t)b * 16 + m) * NN + n) * KK + k] = acc[r];
    }
  }
  for (int t = 0; t < 8; t++) { // row tiles 1..8: w_inte -> intebf (B, C*K, N) bf16 (pre-activation)
    v8f acc = {};
#pragma unroll
    for (int s = 0; s < 8; s++) {
      v16bf a;
#pragma unroll
      for (int j = 0; j < 16; j++) a[j] = wintebf[(size_t)(t * 16 + lm) * 256 + s * 32 + a_koff(j, half)];
      acc = __builtin_amdgcn_wmma_f32_16x16x32_bf16(false, a, false, bfrag[s], (short)0, acc, false, false);
    }
#pragma unroll
    for (int r = 0; r < 8; r++) {
      int o = t * 16 + r + half * 8;
      intebf[(((size_t)b * CC + o) * KK + k) * NN + n] = (bf16_t)acc[r];
    }
  }
}

// ------- small MLP chain per (b,n,k): wx(6->16), wf*wx, all1(16->64), all2(64->128) -------
__global__ void __launch_bounds__(256) mix_kernel(const float* __restrict__ pc, const int* __restrict__ idx,
    const float* __restrict__ wfpre,
    const float* b_fea, const float* s_fea, const float* t_fea,
    const float* w_xyz, const float* b_xyz, const float* s_xyz, const float* t_xyz,
    const float* w_all1, const float* b_all1, const float* s_all1, const float* t_all1,
    const float* w_all2, const float* b_all2, const float* s_all2, const float* t_all2,
    float* __restrict__ ww) {
  int t = blockIdx.x * blockDim.x + threadIdx.x;     // (b*NN+n)*KK + k
  if (t >= BB * NN * KK) return;
  int k = t % KK, bn = t / KK, n = bn % NN, b = bn / NN;
  int nb = idx[t];
  float cx[3], dx[3];
#pragma unroll
  for (int d = 0; d < 3; d++) {
    float c0 = pc[((size_t)b * 3 + d) * NN + n];
    float n0 = pc[((size_t)b * 3 + d) * NN + nb];
    cx[d] = c0; dx[d] = n0 - c0;
  }
  float wv[16];
#pragma unroll
  for (int o = 0; o < 16; o++) {
    float acc = b_xyz[o];
#pragma unroll
    for (int d = 0; d < 3; d++) acc += w_xyz[o * 6 + d] * cx[d] + w_xyz[o * 6 + 3 + d] * dx[d];
    float wx = lrelu(acc * s_xyz[o] + t_xyz[o]);
    float fp = wfpre[(((size_t)b * 16 + o) * NN + n) * KK + k];
    float wf = lrelu((fp + b_fea[o]) * s_fea[o] + t_fea[o]);
    wv[o] = wf * wx;
  }
  float a1[64];
#pragma unroll
  for (int o = 0; o < 64; o++) {
    float acc = 0.f;
#pragma unroll
    for (int c = 0; c < 16; c++) acc += w_all1[o * 16 + c] * wv[c];
    a1[o] = lrelu((acc + b_all1[o]) * s_all1[o] + t_all1[o]);
  }
#pragma unroll 1
  for (int o = 0; o < CC; o++) {
    float acc = 0.f;
#pragma unroll
    for (int c = 0; c < 64; c++) acc += w_all2[o * 64 + c] * a1[c];
    ww[(((size_t)b * CC + o) * NN + n) * KK + k] = lrelu((acc + b_all2[o]) * s_all2[o] + t_all2[o]);
  }
}

// ---------------- softmax over K per (b,c,n), in place ----------------
__global__ void softmax_kernel(float* __restrict__ ww) {
  int t = blockIdx.x * blockDim.x + threadIdx.x;     // b*CC*NN
  if (t >= BB * CC * NN) return;
  float* row = ww + (size_t)t * KK;
  float m = row[0];
  for (int k = 1; k < KK; k++) m = fmaxf(m, row[k]);
  float e[KK]; float s = 0.f;
#pragma unroll
  for (int k = 0; k < KK; k++) { float v = __expf(row[k] - m); e[k] = v; s += v; }
  float inv = 1.f / s;
#pragma unroll
  for (int k = 0; k < KK; k++) row[k] = e[k] * inv;
}

// ------- inte = lrelu(bn(inte_pre)) * ww, in place on bf16 buffer -------
__global__ void inte_elem_kernel(bf16_t* __restrict__ intebf, const float* __restrict__ ww,
                                 const float* b_inte, const float* s_inte, const float* t_inte) {
  size_t t = blockIdx.x * (size_t)blockDim.x + threadIdx.x;  // ((b*CC+o)*KK+k)*NN+n
  if (t >= (size_t)BB * CC * KK * NN) return;
  int n = t % NN; size_t r = t / NN;
  int k = r % KK; r /= KK;
  int o = r % CC; int b = (int)(r / CC);
  float v = (float)intebf[t];
  v = lrelu((v + b_inte[o]) * s_inte[o] + t_inte[o]);
  float w = ww[(((size_t)b * CC + o) * NN + n) * KK + k];
  intebf[t] = (bf16_t)(v * w);
}

// ------- ec = w_c2 (128 x 2560) @ inte (2560 x N), epilogue -> x_ec (WMMA) -------
__global__ void __launch_bounds__(256) c2_wmma_kernel(const bf16_t* __restrict__ w2bf, const bf16_t* __restrict__ intebf,
    const float* b_c2, const float* s_c2, const float* t_c2, const float* s_uc, const float* t_uc,
    bf16_t* __restrict__ xecbf, float* __restrict__ dout) {
  int wave = threadIdx.x >> 5, lane = threadIdx.x & 31;
  int half = lane >> 4, lm = lane & 15;
  int r0 = wave * 16, j0 = blockIdx.x * 16, b = blockIdx.y;
  const bf16_t* ib = intebf + (size_t)b * (CC * KK) * NN;
  v8f acc = {};
  for (int k0 = 0; k0 < CC * KK; k0 += 32) {
    v16bf a, bb;
#pragma unroll
    for (int j = 0; j < 16; j++) {
      a[j]  = w2bf[(size_t)(r0 + lm) * (CC * KK) + k0 + a_koff(j, half)];
      bb[j] = ib[(size_t)(k0 + b_koff(j, half)) * NN + j0 + lm];
    }
    acc = __builtin_amdgcn_wmma_f32_16x16x32_bf16(false, a, false, bb, (short)0, acc, false, false);
  }
  const size_t XOUT = (size_t)BB * CC * NN;
#pragma unroll
  for (int r = 0; r < 8; r++) {
    int o = r0 + r + half * 8, n = j0 + lm;
    float ec = acc[r] + b_c2[o];
    ec = lrelu(ec * s_c2[o] + t_c2[o]);
    float xe = lrelu(ec * s_uc[o] + t_uc[o]);
    xecbf[((size_t)b * CC + o) * NN + n] = (bf16_t)xe;
    dout[XOUT + ((size_t)b * 640 + 512 + o) * NN + n] = xe;   // g_out rows 512..639
  }
}

// ---------------- xs0 = max over n ----------------
__global__ void max_kernel(const float* __restrict__ x, float* __restrict__ xs0) {
  int t = blockIdx.x * blockDim.x + threadIdx.x;     // b*CC + c
  if (t >= BB * CC) return;
  const float* row = x + (size_t)t * NN;
  float m = row[0];
  for (int n = 1; n < NN; n++) m = fmaxf(m, row[n]);
  xs0[t] = m;
}

// ---------------- small FC chain per batch + base for w_add ----------------
__global__ void __launch_bounds__(256) fc_kernel(const float* __restrict__ xs0,
    const float* w_fc1, const float* b_fc1, const float* s_fc1, const float* t_fc1,
    const float* w_fc2, const float* b_fc2, const float* s_fc2, const float* t_fc2,
    const float* w_g, const float* b_g, const float* s_g, const float* t_g,
    const float* w_add, float* __restrict__ gbuf, float* __restrict__ base) {
  __shared__ float s0[CC], s1[CC], s2[CC];
  int b = blockIdx.x, t = threadIdx.x;
  if (t < CC) s0[t] = xs0[b * CC + t];
  __syncthreads();
  if (t < CC) {
    float acc = 0.f;
    for (int c = 0; c < CC; c++) acc += w_fc1[t * CC + c] * s0[c];
    s1[t] = lrelu((acc + b_fc1[t]) * s_fc1[t] + t_fc1[t]);
  }
  __syncthreads();
  if (t < CC) {
    float acc = 0.f;
    for (int c = 0; c < CC; c++) acc += w_fc2[t * CC + c] * s1[c];
    s2[t] = lrelu((acc + b_fc2[t]) * s_fc2[t] + t_fc2[t]);
  }
  __syncthreads();
  for (int o = t; o < 512; o += 256) {
    float acc = 0.f;
    for (int c = 0; c < CC; c++) acc += w_g[o * CC + c] * s2[c];
    gbuf[b * 512 + o] = lrelu((acc + b_g[o]) * s_g[o] + t_g[o]);
  }
  if (t < CC) {
    float acc = 0.f;
    for (int c = 0; c < CC; c++) acc += w_add[t * 2 * CC + c] * s2[c];   // xs3 half of cat
    base[b * CC + t] = acc;
  }
}

// ---------------- broadcast g into g_out rows 0..511 ----------------
__global__ void gbc_kernel(const float* __restrict__ gbuf, float* __restrict__ dout) {
  size_t i = blockIdx.x * (size_t)blockDim.x + threadIdx.x;
  if (i >= (size_t)BB * 512 * NN) return;
  int n = i % NN; size_t r = i / NN;
  int ch = r % 512; int b = (int)(r / 512);
  const size_t XOUT = (size_t)BB * CC * NN;
  dout[XOUT + ((size_t)b * 640 + ch) * NN + n] = gbuf[b * 512 + ch];
}

// ------- x_out = lrelu(bn(base + w_add[:,128:] @ x_ec + b_add)) (WMMA) -------
__global__ void __launch_bounds__(256) add_wmma_kernel(const bf16_t* __restrict__ waddbf, const bf16_t* __restrict__ xecbf,
    const float* base, const float* b_add, const float* s_add, const float* t_add, float* __restrict__ dout) {
  int wave = threadIdx.x >> 5, lane = threadIdx.x & 31;
  int half = lane >> 4, lm = lane & 15;
  int r0 = wave * 16, j0 = blockIdx.x * 16, b = blockIdx.y;
  v8f acc = {};
  for (int k0 = 0; k0 < CC; k0 += 32) {
    v16bf a, bb;
#pragma unroll
    for (int j = 0; j < 16; j++) {
      a[j]  = waddbf[(size_t)(r0 + lm) * CC + k0 + a_koff(j, half)];
      bb[j] = xecbf[((size_t)b * CC + k0 + b_koff(j, half)) * NN + j0 + lm];
    }
    acc = __builtin_amdgcn_wmma_f32_16x16x32_bf16(false, a, false, bb, (short)0, acc, false, false);
  }
#pragma unroll
  for (int r = 0; r < 8; r++) {
    int o = r0 + r + half * 8, n = j0 + lm;
    float v = acc[r] + base[b * CC + o] + b_add[o];
    dout[((size_t)b * CC + o) * NN + n] = lrelu(v * s_add[o] + t_add[o]);
  }
}

extern "C" void kernel_launch(void* const* d_in, const int* in_sizes, int n_in,
                              void* d_out, int out_size, void* d_ws, size_t ws_size,
                              hipStream_t stream) {
  const float* x      = (const float*)d_in[0];
  const float* pc     = (const float*)d_in[1];
  const float* w_fea  = (const float*)d_in[2];
  const float* b_fea  = (const float*)d_in[3];
  const float* s_fea  = (const float*)d_in[4];
  const float* t_fea  = (const float*)d_in[5];
  const float* w_xyz  = (const float*)d_in[6];
  const float* b_xyz  = (const float*)d_in[7];
  const float* s_xyz  = (const float*)d_in[8];
  const float* t_xyz  = (const float*)d_in[9];
  const float* w_all1 = (const float*)d_in[10];
  const float* b_all1 = (const float*)d_in[11];
  const float* s_all1 = (const float*)d_in[12];
  const float* t_all1 = (const float*)d_in[13];
  const float* w_all2 = (const float*)d_in[14];
  const float* b_all2 = (const float*)d_in[15];
  const float* s_all2 = (const float*)d_in[16];
  const float* t_all2 = (const float*)d_in[17];
  const float* w_inte = (const float*)d_in[18];
  const float* b_inte = (const float*)d_in[19];
  const float* s_inte = (const float*)d_in[20];
  const float* t_inte = (const float*)d_in[21];
  const float* w_c2   = (const float*)d_in[22];
  const float* b_c2   = (const float*)d_in[23];
  const float* s_c2   = (const float*)d_in[24];
  const float* t_c2   = (const float*)d_in[25];
  const float* s_uc   = (const float*)d_in[26];
  const float* t_uc   = (const float*)d_in[27];
  const float* w_fc1  = (const float*)d_in[28];
  const float* b_fc1  = (const float*)d_in[29];
  const float* s_fc1  = (const float*)d_in[30];
  const float* t_fc1  = (const float*)d_in[31];
  const float* w_fc2  = (const float*)d_in[32];
  const float* b_fc2  = (const float*)d_in[33];
  const float* s_fc2  = (const float*)d_in[34];
  const float* t_fc2  = (const float*)d_in[35];
  const float* w_g    = (const float*)d_in[36];
  const float* b_g    = (const float*)d_in[37];
  const float* s_g    = (const float*)d_in[38];
  const float* t_g    = (const float*)d_in[39];
  const float* w_add  = (const float*)d_in[40];
  const float* b_add  = (const float*)d_in[41];
  const float* s_add  = (const float*)d_in[42];
  const float* t_add  = (const float*)d_in[43];

  // Workspace layout (bytes). Region0 (84MB) holds dist, then is reused for ww.
  char* ws = (char*)d_ws;
  const size_t OFF_REG0  = 0;                               // max(dist 67MB, ww 84MB)
  const size_t OFF_XBF   = OFF_REG0  + (size_t)BB * CC * NN * KK * 4;
  const size_t OFF_X2    = OFF_XBF   + (size_t)BB * CC * NN * 2;
  const size_t OFF_IDX   = OFF_X2    + (size_t)BB * NN * 4;
  const size_t OFF_WFPRE = OFF_IDX   + (size_t)BB * NN * KK * 4;
  const size_t OFF_INTE  = OFF_WFPRE + (size_t)BB * 16 * NN * KK * 4;
  const size_t OFF_XECBF = OFF_INTE  + (size_t)BB * CC * KK * NN * 2;
  const size_t OFF_W2BF  = OFF_XECBF + (size_t)BB * CC * NN * 2;
  const size_t OFF_WINTB = OFF_W2BF  + (size_t)CC * CC * KK * 2;
  const size_t OFF_WFEAB = OFF_WINTB + (size_t)CC * 2 * CC * 2;
  const size_t OFF_WADDB = OFF_WFEAB + (size_t)16 * 2 * CC * 2;
  const size_t OFF_XS0   = OFF_WADDB + (size_t)CC * CC * 2;
  const size_t OFF_GBUF  = OFF_XS0   + (size_t)BB * CC * 4;
  const size_t OFF_BASE  = OFF_GBUF  + (size_t)BB * 512 * 4;

  float*  reg0    = (float*)(ws + OFF_REG0);     // dist (B,N,N) then ww (B,C,N,K)
  bf16_t* xbf     = (bf16_t*)(ws + OFF_XBF);
  float*  x2      = (float*)(ws + OFF_X2);
  int*    idx     = (int*)(ws + OFF_IDX);
  float*  wfpre   = (float*)(ws + OFF_WFPRE);
  bf16_t* intebf  = (bf16_t*)(ws + OFF_INTE);
  bf16_t* xecbf   = (bf16_t*)(ws + OFF_XECBF);
  bf16_t* w2bf    = (bf16_t*)(ws + OFF_W2BF);
  bf16_t* wintebf = (bf16_t*)(ws + OFF_WINTB);
  bf16_t* wfeabf  = (bf16_t*)(ws + OFF_WFEAB);
  bf16_t* waddbf  = (bf16_t*)(ws + OFF_WADDB);
  float*  xs0     = (float*)(ws + OFF_XS0);
  float*  gbuf    = (float*)(ws + OFF_GBUF);
  float*  base    = (float*)(ws + OFF_BASE);
  float*  dout    = (float*)d_out;

  prep_kernel<<<(BB * NN) / 256, 256, 0, stream>>>(x, xbf, x2);
  convw_kernel<<<1488, 256, 0, stream>>>(w_fea, w_inte, w_c2, w_add, wfeabf, wintebf, w2bf, waddbf);
  dist_kernel<<<dim3(NN / 16 / 8, NN / 16, BB), 256, 0, stream>>>(xbf, x2, reg0);
  topk_kernel<<<(BB * NN) / 256, 256, 0, stream>>>(reg0, idx);
  gather_wmma_kernel<<<dim3((NN * KK / 16) / 8, BB), 256, 0, stream>>>(x, idx, wfeabf, wintebf, wfpre, intebf);
  mix_kernel<<<(BB * NN * KK) / 256, 256, 0, stream>>>(pc, idx, wfpre,
      b_fea, s_fea, t_fea, w_xyz, b_xyz, s_xyz, t_xyz,
      w_all1, b_all1, s_all1, t_all1, w_all2, b_all2, s_all2, t_all2, reg0);
  softmax_kernel<<<(BB * CC * NN) / 256, 256, 0, stream>>>(reg0);
  inte_elem_kernel<<<(int)(((size_t)BB * CC * KK * NN) / 256), 256, 0, stream>>>(intebf, reg0, b_inte, s_inte, t_inte);
  c2_wmma_kernel<<<dim3(NN / 16, BB), 256, 0, stream>>>(w2bf, intebf, b_c2, s_c2, t_c2, s_uc, t_uc, xecbf, dout);
  max_kernel<<<(BB * CC) / 256, 256, 0, stream>>>(x, xs0);
  fc_kernel<<<BB, 256, 0, stream>>>(xs0, w_fc1, b_fc1, s_fc1, t_fc1,
      w_fc2, b_fc2, s_fc2, t_fc2, w_g, b_g, s_g, t_g, w_add, gbuf, base);
  gbc_kernel<<<(int)(((size_t)BB * 512 * NN) / 256), 256, 0, stream>>>(gbuf, dout);
  add_wmma_kernel<<<dim3(NN / 16, BB), 256, 0, stream>>>(waddbf, xecbf, base, b_add, s_add, t_add, dout);
}
